// QuantGaussianBlobNet_40338332844195
// MI455X (gfx1250) — 8-bit matrix operand order repaired and checked by reading, compiled, not run on hardware
//
#include <hip/hip_runtime.h>
#include <stdint.h>


typedef __attribute__((ext_vector_type(8))) int v8i;

#if __has_builtin(__builtin_amdgcn_global_load_async_to_lds_b32)
#define HAVE_ASYNC_G2L 1
#else
#define HAVE_ASYNC_G2L 0
#endif

__device__ __forceinline__ void async_copy_b32(const uint8_t* g, uint8_t* l) {
#if HAVE_ASYNC_G2L
    __builtin_amdgcn_global_load_async_to_lds_b32(
        (__attribute__((address_space(1))) int*)g,
        (__attribute__((address_space(3))) int*)l, 0, 0);
#else
    *(int*)l = *(const int*)g;
#endif
}

__device__ __forceinline__ void wait_async_lds() {
#if HAVE_ASYNC_G2L
#if __has_builtin(__builtin_amdgcn_s_wait_asynccnt)
    __builtin_amdgcn_s_wait_asynccnt(0);
#else
    asm volatile("s_wait_asynccnt 0" ::: "memory");
#endif
#endif
}

static constexpr size_t OFF_SCAL = 0;
static constexpr size_t OFF_Q8   = 256;
static constexpr size_t OFF_B1   = OFF_Q8   + 4194304;
static constexpr size_t OFF_B2   = OFF_B1   + 8388608;
static constexpr size_t OFF_B3   = OFF_B2   + 2097152;
static constexpr size_t OFF_H4   = OFF_B3   + 524288;
static constexpr size_t OFF_SG1  = OFF_H4   + 2048;
static constexpr size_t OFF_SG2  = OFF_SG1  + 1024;
static constexpr size_t OFF_SG3  = OFF_SG2  + 14336;
static constexpr size_t OFF_SGF1 = OFF_SG3  + 57344;
static constexpr size_t OFF_SGF2 = OFF_SGF1 + 4194304;


__global__ void init_slots_kernel(uint32_t* scal) {
    if (threadIdx.x < 10) scal[threadIdx.x] = 0u;
}

__global__ __launch_bounds__(256)
void absmax_kernel(const float* __restrict__ src, int n, uint32_t* slot) {
    __shared__ float red[256];
    float lm = 0.0f;
    for (int i = blockIdx.x * blockDim.x + threadIdx.x; i < n;
         i += gridDim.x * blockDim.x)
        lm = fmaxf(lm, fabsf(src[i]));
    red[threadIdx.x] = lm;
    __syncthreads();
    for (int s = 128; s > 0; s >>= 1) {
        if (threadIdx.x < s)
            red[threadIdx.x] = fmaxf(red[threadIdx.x], red[threadIdx.x + s]);
        __syncthreads();
    }
    if (threadIdx.x == 0)
        atomicMax(slot, __float_as_uint(red[0]));
}

__global__ void quant_x_kernel(const float* __restrict__ x, int n,
                               const uint32_t* __restrict__ scal,
                               int8_t* __restrict__ q) {
    int i = blockIdx.x * blockDim.x + threadIdx.x;
    if (i >= n) return;
    const float s0 = fmaxf(__uint_as_float(scal[0]), 1e-8f) / 127.0f;
    float v = x[i] / s0;
    v = fminf(fmaxf(v, -127.0f), 127.0f);
    q[i] = (int8_t)rintf(v);
}

__global__ void conv_sign_kernel(const float* __restrict__ w, int cout, int cin,
                                 int kpad, int8_t* __restrict__ sgn) {
    int i = blockIdx.x * blockDim.x + threadIdx.x;
    if (i >= cout * kpad) return;
    int n = i / kpad, K = i % kpad;
    int t = K / cin, c = K % cin;
    int8_t v = 0;
    if (t < 27) v = (w[(n * cin + c) * 27 + t] >= 0.0f) ? 1 : -1;
    sgn[i] = v;
}

__global__ void fc1_sign_kernel(const float* __restrict__ w, int8_t* __restrict__ sgn) {
    int i = blockIdx.x * blockDim.x + threadIdx.x;
    if (i >= 128 * 32768) return;
    int n = i >> 15, km = i & 32767;
    int voxel = km >> 6, c = km & 63;
    sgn[i] = (w[n * 32768 + c * 512 + voxel] >= 0.0f) ? 1 : -1;
}

__global__ void fc2_sign_kernel(const float* __restrict__ w, int8_t* __restrict__ sgn) {
    int i = blockIdx.x * blockDim.x + threadIdx.x;
    if (i >= 6 * 128) return;
    sgn[i] = (w[i] >= 0.0f) ? 1 : -1;
}

template <int PASS, int CIN, int COUT, int W>
__global__ __launch_bounds__(512)
void conv_bin_kernel(const uint8_t* __restrict__ qin,
                     const int8_t* __restrict__ sgnw,
                     uint8_t* __restrict__ bout,
                     int* __restrict__ mslot) {
    constexpr int NT = 27;
    constexpr int KPAD = ((NT * CIN + 63) / 64) * 64;
    constexpr int KSTEPS = KPAD / 64;
    constexpr int NTX = W / 16;
    constexpr int WP = W / 2;
    constexpr int LPAD = (CIN < 4) ? 4 : CIN;
    constexpr int ROWB = (LPAD + (W + 1) * CIN + 3) & ~3;
    constexpr int NDWR = (W * CIN) / 4;

    __shared__ __align__(16) uint8_t intile[16 * ROWB];
    __shared__ int red[512];
    __shared__ uint8_t tilebuf[4 * W * COUT];

    const int tid = threadIdx.x;
    const int wave = tid >> 5;
    const int lane = tid & 31;
    const int lhi = (lane >= 16) ? 1 : 0;
    const int l16 = lane & 15;

    int bid = blockIdx.x;
    const int zp = bid % WP; bid /= WP;
    const int yp = bid % WP; bid /= WP;
    const int b = bid;

    for (int i = tid * 4; i < 16 * ROWB; i += 512 * 4)
        *(int*)(intile + i) = 0;
    __syncthreads();
    for (int i = tid; i < 16 * NDWR; i += 512) {
        const int ri = i / NDWR;
        const int dw = i - ri * NDWR;
        const int zz = 2 * zp - 1 + (ri >> 2);
        const int yy = 2 * yp - 1 + (ri & 3);
        if ((unsigned)zz < (unsigned)W && (unsigned)yy < (unsigned)W) {
            const size_t gb =
                ((((size_t)b * W + zz) * W + yy) * W) * CIN + (size_t)dw * 4;
            async_copy_b32(qin + gb, intile + ri * ROWB + LPAD + dw * 4);
        }
    }
    wait_async_lds();
    __syncthreads();

    const int r = wave >> 2;
    const int tmi = wave & 3;
    const int tx = tmi % NTX;
    const int tc = tmi / NTX;
    const int ax = tx * 16 + l16;
    const int rbase = ((r >> 1) * 4 + (r & 1)) * ROWB + LPAD + (ax - 1) * CIN;

    v8i acc = {};
#pragma unroll
    for (int s = 0; s < KSTEPS; ++s) {
        const int k0 = s * 64;
        v8i A, Bm;
#pragma unroll
        for (int v = 0; v < 8; ++v) {
            uint32_t da;
            if constexpr (CIN >= 4) {
                const int kpat = k0 + ((v & 1) << 2) + (((v >> 1) & 1) << 4) +
                                 ((v >> 2) << 5);
                const int t = kpat / CIN;
                if (t < NT) {
                    const int c = kpat % CIN + lhi * 8;
                    const int dz = t / 9, dyq = (t / 3) % 3, dx = t % 3;
                    da = *(const uint32_t*)(intile + rbase +
                                            (dz * 4 + dyq) * ROWB + dx * CIN + c);
                } else {
                    da = 0u;
                }
            } else {
                da = 0u;
#pragma unroll
                for (int j = 0; j < 4; ++j) {
                    const int t = k0 + ((v & 1) << 2) + (((v >> 1) & 1) << 4) +
                                  ((v >> 2) << 5) + (lhi << 3) + j;
                    if (t < NT) {
                        const int dz = t / 9, dyq = (t / 3) % 3, dx = t % 3;
                        uint32_t byte =
                            intile[rbase + (dz * 4 + dyq) * ROWB + dx];
                        da |= byte << (8 * j);
                    }
                }
            }
            A[v] = (int)da;

            const int kbB = k0 + ((v & 1) << 2) + (((v >> 1) & 1) << 4) + ((v >> 2) << 5) + (lhi << 3);
            Bm[v] = *(const int*)(sgnw + (size_t)(tc * 16 + l16) * KPAD + kbB);
        }
        acc = __builtin_amdgcn_wmma_i32_16x16x64_iu8(
            (CIN == 1), A, true, Bm, acc, false, false);
    }

    if constexpr (PASS == 0) {
        int lm = 0;
#pragma unroll
        for (int v = 0; v < 8; ++v) lm = max(lm, acc[v]);
        red[tid] = lm;
        __syncthreads();
        for (int s = 256; s > 0; s >>= 1) {
            if (tid < s) red[tid] = max(red[tid], red[tid + s]);
            __syncthreads();
        }
        if (tid == 0) atomicMax(mslot, red[0]);
    } else {
        const int M = *mslot;
#pragma unroll
        for (int v = 0; v < 8; ++v) {
            const int mrow = v + lhi * 8;
            const int xx = tx * 16 + mrow;
            const int ch = tc * 16 + l16;
            const int val = acc[v];
            tilebuf[(r * W + xx) * COUT + ch] = (val > 0 && 2 * val > M) ? 1 : 0;
        }
        __syncthreads();
        const int xp = tid / COUT;
        const int ch = tid % COUT;
        uint8_t p = 0;
#pragma unroll
        for (int rr = 0; rr < 4; ++rr)
#pragma unroll
            for (int xo = 0; xo < 2; ++xo)
                p |= tilebuf[(rr * W + (2 * xp + xo)) * COUT + ch];
        const size_t oidx =
            ((((size_t)b * WP + zp) * WP + yp) * WP + xp) * COUT + ch;
        bout[oidx] = p;
    }
}

__global__ __launch_bounds__(256)
void fc1_kernel(const uint8_t* __restrict__ h3,
                const int8_t* __restrict__ sgnw,
                uint8_t* __restrict__ h4,
                int* __restrict__ m4slot) {
    __shared__ int red[256];
    __shared__ int Mg;
    constexpr int K = 32768;
    const int tid = threadIdx.x;
    const int wave = tid >> 5;
    const int lane = tid & 31;
    const int lhi = (lane >= 16) ? 1 : 0;
    const int l16 = lane & 15;

    v8i acc = {};
    for (int s = 0; s < K / 64; ++s) {
        const int k0 = s * 64;
        v8i A, Bm;
#pragma unroll
        for (int v = 0; v < 8; ++v) {
            const int kbA = k0 + ((v & 1) << 2) + (((v >> 1) & 1) << 4) +
                            ((v >> 2) << 5) + (lhi << 3);
            A[v] = *(const int*)(h3 + (size_t)l16 * K + kbA);
            const int kbB = k0 + ((v & 1) << 2) + (((v >> 1) & 1) << 4) + ((v >> 2) << 5) + (lhi << 3);
            const int n = wave * 16 + l16;
            Bm[v] = *(const int*)(sgnw + (size_t)n * K + kbB);
        }
        acc = __builtin_amdgcn_wmma_i32_16x16x64_iu8(false, A, true, Bm, acc,
                                                     false, false);
    }
    int lm = 0;
#pragma unroll
    for (int v = 0; v < 8; ++v) lm = max(lm, acc[v]);
    red[tid] = lm;
    __syncthreads();
    for (int s = 128; s > 0; s >>= 1) {
        if (tid < s) red[tid] = max(red[tid], red[tid + s]);
        __syncthreads();
    }
    if (tid == 0) { Mg = red[0]; *m4slot = red[0]; }
    __syncthreads();
    const int M = Mg;
#pragma unroll
    for (int v = 0; v < 8; ++v) {
        const int batch = v + lhi * 8;
        const int col = wave * 16 + l16;
        const int val = acc[v];
        h4[batch * 128 + col] = (val > 0 && 2 * val > M) ? 1 : 0;
    }
}

__global__ void fc2_kernel(const uint8_t* __restrict__ h4,
                           const int8_t* __restrict__ sgn2,
                           const uint32_t* __restrict__ scal,
                           float* __restrict__ out) {
    const int tid = threadIdx.x;
    if (tid >= 96) return;
    const int b = tid / 6, j = tid % 6;
    int acc = 0;
    for (int k = 0; k < 128; ++k)
        acc += (int)h4[b * 128 + k] * (int)sgn2[j * 128 + k];
    const float s0  = fmaxf(__uint_as_float(scal[0]), 1e-8f) / 127.0f;
    const float sw1 = fmaxf(__uint_as_float(scal[1]), 1e-8f);
    const float sw2 = fmaxf(__uint_as_float(scal[2]), 1e-8f);
    const float sw3 = fmaxf(__uint_as_float(scal[3]), 1e-8f);
    const float sf1 = fmaxf(__uint_as_float(scal[4]), 1e-8f);
    const float sf2 = fmaxf(__uint_as_float(scal[5]), 1e-8f);
    const float a1 = fmaxf(s0 * sw1 * (float)(int)scal[6], 1e-8f);
    const float a2 = fmaxf(a1 * sw2 * (float)(int)scal[7], 1e-8f);
    const float a3 = fmaxf(a2 * sw3 * (float)(int)scal[8], 1e-8f);
    const float a4 = fmaxf(a3 * sf1 * (float)(int)scal[9], 1e-8f);
    out[tid] = (float)acc * a4 * sf2;
}

extern "C" void kernel_launch(void* const* d_in, const int* in_sizes, int n_in,
                              void* d_out, int out_size, void* d_ws, size_t ws_size,
                              hipStream_t stream) {
    const float* x    = (const float*)d_in[0];
    const float* w1   = (const float*)d_in[1];
    const float* w2   = (const float*)d_in[2];
    const float* w3   = (const float*)d_in[3];
    const float* wfc1 = (const float*)d_in[4];
    const float* wfc2 = (const float*)d_in[5];
    float* out = (float*)d_out;

    uint8_t* ws = (uint8_t*)d_ws;
    uint32_t* scal = (uint32_t*)(ws + OFF_SCAL);
    int8_t*  q8   = (int8_t*)(ws + OFF_Q8);
    uint8_t* b1   = ws + OFF_B1;
    uint8_t* b2   = ws + OFF_B2;
    uint8_t* b3   = ws + OFF_B3;
    uint8_t* h4   = ws + OFF_H4;
    int8_t*  sg1  = (int8_t*)(ws + OFF_SG1);
    int8_t*  sg2  = (int8_t*)(ws + OFF_SG2);
    int8_t*  sg3  = (int8_t*)(ws + OFF_SG3);
    int8_t*  sgf1 = (int8_t*)(ws + OFF_SGF1);
    int8_t*  sgf2 = (int8_t*)(ws + OFF_SGF2);

    const int NX = 16 * 64 * 64 * 64;

    init_slots_kernel<<<1, 32, 0, stream>>>(scal);

    absmax_kernel<<<1024, 256, 0, stream>>>(x, NX, scal + 0);
    absmax_kernel<<<2, 256, 0, stream>>>(w1, 16 * 27, scal + 1);
    absmax_kernel<<<16, 256, 0, stream>>>(w2, 32 * 16 * 27, scal + 2);
    absmax_kernel<<<32, 256, 0, stream>>>(w3, 64 * 32 * 27, scal + 3);
    absmax_kernel<<<1024, 256, 0, stream>>>(wfc1, 128 * 32768, scal + 4);
    absmax_kernel<<<2, 256, 0, stream>>>(wfc2, 6 * 128, scal + 5);

    conv_sign_kernel<<<(16 * 64 + 255) / 256, 256, 0, stream>>>(w1, 16, 1, 64, sg1);
    conv_sign_kernel<<<(32 * 448 + 255) / 256, 256, 0, stream>>>(w2, 32, 16, 448, sg2);
    conv_sign_kernel<<<(64 * 896 + 255) / 256, 256, 0, stream>>>(w3, 64, 32, 896, sg3);
    fc1_sign_kernel<<<(128 * 32768) / 256, 256, 0, stream>>>(wfc1, sgf1);
    fc2_sign_kernel<<<3, 256, 0, stream>>>(wfc2, sgf2);

    quant_x_kernel<<<(NX + 255) / 256, 256, 0, stream>>>(x, NX, scal, q8);

    conv_bin_kernel<0, 1, 16, 64><<<16 * 32 * 32, 512, 0, stream>>>(
        (const uint8_t*)q8, sg1, b1, (int*)(scal + 6));
    conv_bin_kernel<1, 1, 16, 64><<<16 * 32 * 32, 512, 0, stream>>>(
        (const uint8_t*)q8, sg1, b1, (int*)(scal + 6));

    conv_bin_kernel<0, 16, 32, 32><<<16 * 16 * 16, 512, 0, stream>>>(
        b1, sg2, b2, (int*)(scal + 7));
    conv_bin_kernel<1, 16, 32, 32><<<16 * 16 * 16, 512, 0, stream>>>(
        b1, sg2, b2, (int*)(scal + 7));

    conv_bin_kernel<0, 32, 64, 16><<<16 * 8 * 8, 512, 0, stream>>>(
        b2, sg3, b3, (int*)(scal + 8));
    conv_bin_kernel<1, 32, 64, 16><<<16 * 8 * 8, 512, 0, stream>>>(
        b2, sg3, b3, (int*)(scal + 8));

    fc1_kernel<<<1, 256, 0, stream>>>(b3, sgf1, h4, (int*)(scal + 9));
    fc2_kernel<<<1, 128, 0, stream>>>(h4, sgf2, scal, out);
}
